// SAGPooling_53747220742801
// MI455X (gfx1250) — matrix operand order repaired and checked by reading, not compiled, not run on hardware
//
#include <hip/hip_runtime.h>

// ---------------- problem constants ----------------
constexpr int BATCH  = 64;
constexpr int N_PER  = 1024;
constexpr int IN_CH  = 256;
constexpr int HEADS  = 2;
constexpr int OUT_CH = 128;
constexpr int NNODE  = BATCH * N_PER;       // 65536
constexpr int NEDGE  = 524288;              // B*N_PER*DEG
constexpr int KKEEP  = 512;                 // nodes kept per graph
constexpr int BK     = BATCH * KKEEP;       // 32768 pooled nodes
constexpr float NEG_SLOPE = 0.2f;

typedef __attribute__((ext_vector_type(16))) __bf16 v16bf;
typedef __attribute__((ext_vector_type(8)))  float  v8f;

// ---------------- helpers ----------------
__device__ __forceinline__ unsigned enc_f(float f) {
    unsigned u = __float_as_uint(f);
    return (u & 0x80000000u) ? ~u : (u | 0x80000000u);
}
__device__ __forceinline__ float dec_f(unsigned u) {
    return (u & 0x80000000u) ? __uint_as_float(u & 0x7FFFFFFFu)
                             : __uint_as_float(~u);
}
__device__ __forceinline__ float leaky(float x) {
    return x > 0.f ? x : NEG_SLOPE * x;
}
// pack two f32 -> packed bf16 pair (lo in [15:0]), round-to-nearest-even
__device__ __forceinline__ unsigned pk_bf16(float lo, float hi) {
    unsigned a = __float_as_uint(lo), b = __float_as_uint(hi);
    a += 0x7FFFu + ((a >> 16) & 1u);
    b += 0x7FFFu + ((b >> 16) & 1u);
    return (a >> 16) | (b & 0xFFFF0000u);
}

// ---------------- 0: init ----------------
__global__ void k_init(unsigned* m_enc, float* denom, int* indeg, int* new_idx) {
    int n = blockIdx.x * blockDim.x + threadIdx.x;
    if (n >= NNODE) return;
    m_enc[2*n] = 0u; m_enc[2*n+1] = 0u;
    denom[2*n] = 0.f; denom[2*n+1] = 0.f;
    indeg[n] = 0;
    new_idx[n] = -1;
}

// ---------------- 1: hw = x @ W  (bf16 WMMA, fragment-swizzled LDS) ----------------
// LDS layout: per 16x32 (A) / 32x16 (B) tile: 32 lanes x 32 bytes, each lane's
// 16 bf16 fragment elements contiguous -> operand load = one 32B LDS read.
__global__ __launch_bounds__(256) void k_gemm(const float* __restrict__ x,
                                              const float* __restrict__ W,
                                              float* __restrict__ hw) {
    __shared__ unsigned AsSw[8 * 256];   // 8 row-tiles  * 256 u32 (1KB each)
    __shared__ unsigned BsSw[8 * 256];   // 8 col-tiles
    const int tid  = threadIdx.x;
    const int wave = tid >> 5;
    const int lane = tid & 31;
    const int m16  = lane & 15;
    const int g    = lane >> 4;
    const int rowBase = blockIdx.x * 128;
    const int colBase = blockIdx.y * 128;

    v8f acc[8];
    for (int t = 0; t < 8; ++t)
        for (int r = 0; r < 8; ++r) acc[t][r] = 0.f;

    for (int k0 = 0; k0 < IN_CH; k0 += 32) {
        if (k0 + 32 < IN_CH) {   // prefetch next K slab -> global_prefetch_b8
            __builtin_prefetch(&x[(size_t)(rowBase + (tid >> 1)) * IN_CH + k0 + 32], 0, 1);
            __builtin_prefetch(&W[(size_t)(k0 + 32 + (tid >> 3)) * IN_CH + colBase], 0, 1);
        }
        // Stage A (128x32) and B (32x128) as packed bf16 pairs, pre-swizzled.
        for (int i = 0; i < 8; ++i) {
            int p = i * 256 + tid;                 // pair index 0..2047
            // ---- A: pair along k (contiguous in x -> float2 load) ----
            {
                int r  = p >> 4;                   // 0..127
                int k  = (p & 15) * 2;             // even k
                float2 v = *reinterpret_cast<const float2*>(
                    &x[(size_t)(rowBase + r) * IN_CH + k0 + k]);
                int blk = k >> 3;                  // 0..3
                int gA  = blk & 1;
                int jA  = (k & 7) + (blk >> 1) * 8;
                int lA  = gA * 16 + (r & 15);
                AsSw[(r >> 4) * 256 + lA * 8 + (jA >> 1)] = pk_bf16(v.x, v.y);
            }
            // ---- B: pair along k (strided in W -> two coalesced dword loads) ----
            {
                int n  = p & 127;                  // 0..127
                int kb = (p >> 7) * 2;             // even k
                float blo = W[(size_t)(k0 + kb)     * IN_CH + colBase + n];
                float bhi = W[(size_t)(k0 + kb + 1) * IN_CH + colBase + n];
                int gB = (kb >> 3) & 1;
                int jB = (kb & 7) + (kb >> 4) * 8;
                int lB = gB * 16 + (n & 15);
                BsSw[(n >> 4) * 256 + lB * 8 + (jB >> 1)] = pk_bf16(blo, bhi);
            }
        }
        __syncthreads();

        v16bf a = *reinterpret_cast<const v16bf*>(&AsSw[wave * 256 + lane * 8]);
        for (int t = 0; t < 8; ++t) {
            v16bf b = *reinterpret_cast<const v16bf*>(&BsSw[t * 256 + lane * 8]);
            acc[t] = __builtin_amdgcn_wmma_f32_16x16x32_bf16(
                false, a, false, b, (short)0, acc[t], false, false);
        }
        __syncthreads();
    }
    // C/D layout: VGPR r -> row r + 8g, col = m16
    for (int t = 0; t < 8; ++t)
        for (int r = 0; r < 8; ++r)
            hw[(size_t)(rowBase + wave * 16 + r + 8 * g) * IN_CH + colBase + t * 16 + m16] =
                acc[t][r];
}

// ---------------- 2: attention coefficients ----------------
__global__ __launch_bounds__(256) void k_attn(const float* __restrict__ hw,
                                              const float* __restrict__ att_src,
                                              const float* __restrict__ att_dst,
                                              float* __restrict__ a_s,
                                              float* __restrict__ a_d) {
    int warp = threadIdx.x >> 5, lane = threadIdx.x & 31;
    int n = blockIdx.x * 8 + warp;
    float s0 = 0, s1 = 0, d0 = 0, d1 = 0;
    for (int j = 0; j < 8; ++j) {
        int c = lane + 32 * j;
        float v = hw[(size_t)n * IN_CH + c];
        float as = att_src[c], ad = att_dst[c];   // [H,OUT_CH] flat == channel c
        if (j < 4) { s0 += v * as; d0 += v * ad; }
        else       { s1 += v * as; d1 += v * ad; }
    }
    for (int off = 16; off; off >>= 1) {
        s0 += __shfl_xor(s0, off); s1 += __shfl_xor(s1, off);
        d0 += __shfl_xor(d0, off); d1 += __shfl_xor(d1, off);
    }
    if (lane == 0) {
        a_s[2*n] = s0; a_s[2*n+1] = s1;
        a_d[2*n] = d0; a_d[2*n+1] = d1;
    }
}

// ---------------- 3: segment max (edges + self loops) ----------------
__global__ void k_max(const long long* __restrict__ ei,
                      const float* __restrict__ a_s, const float* __restrict__ a_d,
                      unsigned* __restrict__ m_enc) {
    int t = blockIdx.x * blockDim.x + threadIdx.x;
    if (t >= NEDGE + NNODE) return;
    int s, d;
    if (t < NEDGE) { s = (int)ei[t]; d = (int)ei[NEDGE + t]; }
    else           { s = d = t - NEDGE; }
    float e0 = leaky(a_s[2*s]   + a_d[2*d]);
    float e1 = leaky(a_s[2*s+1] + a_d[2*d+1]);
    atomicMax(&m_enc[2*d],   enc_f(e0));
    atomicMax(&m_enc[2*d+1], enc_f(e1));
}

// ---------------- 4: exp-sum + in-degree ----------------
__global__ void k_expsum(const long long* __restrict__ ei,
                         const float* __restrict__ a_s, const float* __restrict__ a_d,
                         const unsigned* __restrict__ m_enc,
                         float* __restrict__ denom, int* __restrict__ indeg) {
    int t = blockIdx.x * blockDim.x + threadIdx.x;
    if (t >= NEDGE + NNODE) return;
    int s, d;
    if (t < NEDGE) { s = (int)ei[t]; d = (int)ei[NEDGE + t]; }
    else           { s = d = t - NEDGE; }
    float e0 = leaky(a_s[2*s]   + a_d[2*d]);
    float e1 = leaky(a_s[2*s+1] + a_d[2*d+1]);
    atomicAdd(&denom[2*d],   expf(e0 - dec_f(m_enc[2*d])));
    atomicAdd(&denom[2*d+1], expf(e1 - dec_f(m_enc[2*d+1])));
    if (t < NEDGE) atomicAdd(&indeg[d], 1);
}

// ---------------- 5: exclusive scan of in-degrees (single block) ----------------
__global__ __launch_bounds__(1024) void k_scan(const int* __restrict__ indeg,
                                               int* __restrict__ row_start,
                                               int* __restrict__ cursor) {
    __shared__ int buf[1024];
    __shared__ int carry;
    int tid = threadIdx.x;
    if (tid == 0) carry = 0;
    __syncthreads();
    for (int chunk = 0; chunk < NNODE / 1024; ++chunk) {
        int i = chunk * 1024 + tid;
        int v = indeg[i];
        buf[tid] = v;
        __syncthreads();
        for (int off = 1; off < 1024; off <<= 1) {
            int t = (tid >= off) ? buf[tid - off] : 0;
            __syncthreads();
            buf[tid] += t;
            __syncthreads();
        }
        int base = carry;
        int excl = base + buf[tid] - v;
        row_start[i] = excl;
        cursor[i]    = excl;
        __syncthreads();
        if (tid == 0) carry = base + buf[1023];
        __syncthreads();
    }
}

// ---------------- 6: fill dst-CSR with (src, exp) ----------------
__global__ void k_fill(const long long* __restrict__ ei,
                       const float* __restrict__ a_s, const float* __restrict__ a_d,
                       const unsigned* __restrict__ m_enc,
                       int* __restrict__ cursor, int* __restrict__ csr_src,
                       float* __restrict__ csr_ex) {
    int t = blockIdx.x * blockDim.x + threadIdx.x;
    if (t >= NEDGE) return;
    int s = (int)ei[t], d = (int)ei[NEDGE + t];
    float e0 = leaky(a_s[2*s]   + a_d[2*d]);
    float e1 = leaky(a_s[2*s+1] + a_d[2*d+1]);
    float ex0 = expf(e0 - dec_f(m_enc[2*d]));
    float ex1 = expf(e1 - dec_f(m_enc[2*d+1]));
    int pos = atomicAdd(&cursor[d], 1);
    csr_src[pos]   = s;
    csr_ex[2*pos]   = ex0;
    csr_ex[2*pos+1] = ex1;
}

// ---------------- 7: fused aggregate + bias + relu + score ----------------
__global__ __launch_bounds__(256) void k_gather(const float* __restrict__ hw,
        const int* __restrict__ row_start, const int* __restrict__ indeg,
        const int* __restrict__ csr_src, const float* __restrict__ csr_ex,
        const float* __restrict__ a_s, const float* __restrict__ a_d,
        const unsigned* __restrict__ m_enc, const float* __restrict__ denom,
        const float* __restrict__ bias, const float* __restrict__ aw,
        float* __restrict__ h, float* __restrict__ scores) {
    int warp = threadIdx.x >> 5, lane = threadIdx.x & 31;
    int n = blockIdx.x * 8 + warp;
    float m0 = dec_f(m_enc[2*n]), m1 = dec_f(m_enc[2*n+1]);
    float inv0 = 1.f / (denom[2*n]   + 1e-16f);
    float inv1 = 1.f / (denom[2*n+1] + 1e-16f);
    float acc[8];
    for (int j = 0; j < 8; ++j) acc[j] = 0.f;

    int rs = row_start[n], re = rs + indeg[n];
    for (int i = rs; i < re; ++i) {
        int s = csr_src[i];
        float al0 = csr_ex[2*i]   * inv0;
        float al1 = csr_ex[2*i+1] * inv1;
        const float* row = &hw[(size_t)s * IN_CH];
        for (int j = 0; j < 8; ++j)
            acc[j] += ((j < 4) ? al0 : al1) * row[lane + 32 * j];
    }
    // self loop
    {
        float e0 = leaky(a_s[2*n]   + a_d[2*n]);
        float e1 = leaky(a_s[2*n+1] + a_d[2*n+1]);
        float al0 = expf(e0 - m0) * inv0;
        float al1 = expf(e1 - m1) * inv1;
        const float* row = &hw[(size_t)n * IN_CH];
        for (int j = 0; j < 8; ++j)
            acc[j] += ((j < 4) ? al0 : al1) * row[lane + 32 * j];
    }
    float p = 0.f;
    for (int j = 0; j < 8; ++j) {
        int c = lane + 32 * j;
        float v = acc[j] + bias[c];
        v = v > 0.f ? v : 0.f;                 // relu
        h[(size_t)n * IN_CH + c] = v;
        p += v * aw[c];                        // fused score GEMV
    }
    for (int off = 16; off; off >>= 1) p += __shfl_xor(p, off);
    if (lane == 0) scores[n] = tanhf(p);
}

// ---------------- 8: per-graph top-k via bitonic sort ----------------
__global__ __launch_bounds__(1024) void k_topk(const float* __restrict__ scores,
                                               float* __restrict__ out_perm,
                                               float* __restrict__ out_batch,
                                               int* __restrict__ perm_i,
                                               int* __restrict__ new_idx) {
    __shared__ float key[1024];
    __shared__ int   val[1024];
    int b = blockIdx.x, tid = threadIdx.x;
    key[tid] = scores[b * N_PER + tid];
    val[tid] = tid;
    __syncthreads();
    for (int k = 2; k <= 1024; k <<= 1) {
        for (int j = k >> 1; j > 0; j >>= 1) {
            int ixj = tid ^ j;
            if (ixj > tid) {
                float ka = key[tid], kb = key[ixj];
                int   ia = val[tid], ib = val[ixj];
                // "b before a" in final order (score desc, idx asc on tie)
                bool before_b = (kb > ka) || (kb == ka && ib < ia);
                bool asc = ((tid & k) == 0);
                if (asc ? before_b : !before_b) {
                    key[tid] = kb; key[ixj] = ka;
                    val[tid] = ib; val[ixj] = ia;
                }
            }
            __syncthreads();
        }
    }
    if (tid < KKEEP) {
        int gi = b * N_PER + val[tid];
        int p  = b * KKEEP + tid;
        perm_i[p]   = gi;
        new_idx[gi] = p;
        out_perm[p]  = (float)gi;
        out_batch[p] = (float)b;
    }
}

// ---------------- 9: gather pooled node features ----------------
__global__ __launch_bounds__(256) void k_pool(const float* __restrict__ h,
                                              const int* __restrict__ perm_i,
                                              float* __restrict__ out_x) {
    int row = blockIdx.x, c = threadIdx.x;
    out_x[(size_t)row * IN_CH + c] = h[(size_t)perm_i[row] * IN_CH + c];
}

// ---------------- 10: remap surviving edges ----------------
__global__ void k_edges(const long long* __restrict__ ei,
                        const int* __restrict__ new_idx,
                        float* __restrict__ out_e) {
    int t = blockIdx.x * blockDim.x + threadIdx.x;
    if (t >= NEDGE) return;
    int s = (int)ei[t], d = (int)ei[NEDGE + t];
    int sn = new_idx[s], dn = new_idx[d];
    bool valid = (sn >= 0) && (dn >= 0);
    out_e[t]         = valid ? (float)sn : -1.f;
    out_e[NEDGE + t] = valid ? (float)dn : -1.f;
}

// ---------------- launch ----------------
extern "C" void kernel_launch(void* const* d_in, const int* in_sizes, int n_in,
                              void* d_out, int out_size, void* d_ws, size_t ws_size,
                              hipStream_t stream) {
    const float*     x   = (const float*)d_in[0];
    const long long* ei  = (const long long*)d_in[1];
    // d_in[2] = batch (recomputed analytically)
    const float*     W   = (const float*)d_in[3];
    const float*     asv = (const float*)d_in[4];
    const float*     adv = (const float*)d_in[5];
    const float*     bia = (const float*)d_in[6];
    const float*     aw  = (const float*)d_in[7];

    char* ws = (char*)d_ws;
    size_t off = 0;
    auto take = [&](size_t bytes) { char* p = ws + off; off += (bytes + 255) & ~255ull; return p; };
    float*    hw       = (float*)   take((size_t)NNODE * IN_CH * 4);
    float*    h        = (float*)   take((size_t)NNODE * IN_CH * 4);
    float*    a_s      = (float*)   take((size_t)NNODE * 2 * 4);
    float*    a_d      = (float*)   take((size_t)NNODE * 2 * 4);
    unsigned* m_enc    = (unsigned*)take((size_t)NNODE * 2 * 4);
    float*    denom    = (float*)   take((size_t)NNODE * 2 * 4);
    int*      indeg    = (int*)     take((size_t)NNODE * 4);
    int*      row_st   = (int*)     take((size_t)NNODE * 4);
    int*      cursor   = (int*)     take((size_t)NNODE * 4);
    int*      csr_src  = (int*)     take((size_t)NEDGE * 4);
    float*    csr_ex   = (float*)   take((size_t)NEDGE * 2 * 4);
    float*    scores   = (float*)   take((size_t)NNODE * 4);
    int*      perm_i   = (int*)     take((size_t)BK * 4);
    int*      new_idx  = (int*)     take((size_t)NNODE * 4);
    (void)ws_size; (void)in_sizes; (void)n_in; (void)out_size;

    float* out_x = (float*)d_out;                   // [BK, 256]
    float* out_e = out_x + (size_t)BK * IN_CH;      // [2, NEDGE]
    float* out_b = out_e + 2 * (size_t)NEDGE;       // [BK]
    float* out_p = out_b + BK;                      // [BK]

    k_init  <<<NNODE / 256, 256, 0, stream>>>(m_enc, denom, indeg, new_idx);
    k_gemm  <<<dim3(NNODE / 128, IN_CH / 128), 256, 0, stream>>>(x, W, hw);
    k_attn  <<<NNODE / 8, 256, 0, stream>>>(hw, asv, adv, a_s, a_d);
    k_max   <<<(NEDGE + NNODE) / 256, 256, 0, stream>>>(ei, a_s, a_d, m_enc);
    k_expsum<<<(NEDGE + NNODE) / 256, 256, 0, stream>>>(ei, a_s, a_d, m_enc, denom, indeg);
    k_scan  <<<1, 1024, 0, stream>>>(indeg, row_st, cursor);
    k_fill  <<<NEDGE / 256, 256, 0, stream>>>(ei, a_s, a_d, m_enc, cursor, csr_src, csr_ex);
    k_gather<<<NNODE / 8, 256, 0, stream>>>(hw, row_st, indeg, csr_src, csr_ex,
                                            a_s, a_d, m_enc, denom, bia, aw, h, scores);
    k_topk  <<<BATCH, 1024, 0, stream>>>(scores, out_p, out_b, perm_i, new_idx);
    k_pool  <<<BK, 256, 0, stream>>>(h, perm_i, out_x);
    k_edges <<<NEDGE / 256, 256, 0, stream>>>(ei, new_idx, out_e);
}
